// MLAAttention_13417477833164
// MI455X (gfx1250) — matrix operand order repaired and checked by reading, not compiled, not run on hardware
//
#include <hip/hip_runtime.h>

#define T_SEQ   2048
#define HIDDEN  5120
#define NHEADS  32
#define Q_LORA  1536
#define KV_LORA 512
#define NOPE    128
#define ROPE_D  64
#define V_DIM   128
#define QKD     192   // NOPE + ROPE

typedef unsigned int   u32;
typedef unsigned short u16;

typedef __attribute__((ext_vector_type(16))) __bf16 v16bf;
typedef __attribute__((ext_vector_type(8)))  float  v8f;
typedef int v4i_ __attribute__((vector_size(16)));   // matches builtin param type

union Frag {
    v16bf v;
    u32   u[8];
};

// ---- gfx1250 async global->LDS copy (ASYNCcnt path), with safe fallback ----
#if defined(__has_builtin)
#  if __has_builtin(__builtin_amdgcn_global_load_async_to_lds_b128)
#    define HAVE_ASYNC_LDS 1
#  endif
#endif
#ifndef HAVE_ASYNC_LDS
#  define HAVE_ASYNC_LDS 0
#endif

#if HAVE_ASYNC_LDS
typedef __attribute__((address_space(1))) v4i_ gv4i_t;
typedef __attribute__((address_space(3))) v4i_ lv4i_t;
__device__ __forceinline__ void async_copy16(const u16* g, u16* l) {
    __builtin_amdgcn_global_load_async_to_lds_b128(
        (gv4i_t*)(u16*)g,
        (lv4i_t*)l,
        0, 0);
}
__device__ __forceinline__ void wait_async0() {
#  if __has_builtin(__builtin_amdgcn_s_wait_asynccnt)
    __builtin_amdgcn_s_wait_asynccnt(0);
#  else
    asm volatile("s_wait_asynccnt 0x0" ::: "memory");
#  endif
}
#endif

__device__ __forceinline__ u16 f2bf(float f) {
    union { float f; u32 u; } x; x.f = f;
    u32 u = x.u;
    u += 0x7FFFu + ((u >> 16) & 1u);   // round-to-nearest-even
    return (u16)(u >> 16);
}

__device__ __forceinline__ v8f vzero8() {
    v8f z;
#pragma unroll
    for (int i = 0; i < 8; ++i) z[i] = 0.0f;
    return z;
}

__device__ __forceinline__ v8f wmma_bf16(v16bf a, v16bf b, v8f c) {
    // D = A(16x32 bf16) x B(32x16 bf16) + C(16x16 f32)
    return __builtin_amdgcn_wmma_f32_16x16x32_bf16(false, a, false, b,
                                                   (short)0, c, false, false);
}

// ---------------------------------------------------------------------------
// fp32 -> bf16 elementwise convert
// ---------------------------------------------------------------------------
__global__ void k_cvt_bf16(const float* __restrict__ in, u16* __restrict__ out, int n) {
    int i = blockIdx.x * 256 + threadIdx.x;
    if (i < n) out[i] = f2bf(in[i]);
}

// ---------------------------------------------------------------------------
// fp32 [K][N] -> bf16 transposed [N][K]  (weights; makes GEMM B k-contiguous)
// ---------------------------------------------------------------------------
__global__ void k_cvt_bf16_t(const float* __restrict__ in, u16* __restrict__ out,
                             int K, int N) {
    int idx = blockIdx.x * 256 + threadIdx.x;
    if (idx >= K * N) return;
    int n = idx / K, k = idx % K;
    out[idx] = f2bf(in[(size_t)k * N + n]);
}

// ---------------------------------------------------------------------------
// RMSNorm over `cols` of each row (input fp32 strided), output bf16
// ---------------------------------------------------------------------------
__global__ __launch_bounds__(256) void k_rmsnorm_bf16(
    const float* __restrict__ in, int in_stride,
    const float* __restrict__ w, int cols,
    u16* __restrict__ out, int out_stride) {
    int row = blockIdx.x;
    const float* x = in + (size_t)row * in_stride;
    __shared__ float red[256];
    float ss = 0.0f;
    for (int c = threadIdx.x; c < cols; c += 256) { float v = x[c]; ss += v * v; }
    red[threadIdx.x] = ss;
    __syncthreads();
    for (int s = 128; s > 0; s >>= 1) {
        if ((int)threadIdx.x < s) red[threadIdx.x] += red[threadIdx.x + s];
        __syncthreads();
    }
    float scale = rsqrtf(red[0] / (float)cols + 1e-6f);
    for (int c = threadIdx.x; c < cols; c += 256)
        out[(size_t)row * out_stride + c] = f2bf(x[c] * scale * w[c]);
}

// ---------------------------------------------------------------------------
// C (fp32, MxN) = A (bf16, MxK rowmajor) * Bt (bf16, NxK rowmajor, i.e. B^T)
// block tile 128x64, 8 waves, each wave 32x32 (2x2 WMMA tiles), k-step 32.
// LDS row stride 40 u16 = 80B -> every 16B chunk is 16B-aligned (async b128).
// M % 128 == 0, N % 64 == 0, K % 32 == 0 (true for all call sites).
// ---------------------------------------------------------------------------
__global__ __launch_bounds__(256) void k_gemm_bf16(
    const u16* __restrict__ A, const u16* __restrict__ Bt, float* __restrict__ C,
    int M, int N, int K) {
    __shared__ __align__(16) u16 As[128][40];   // [row][k]
    __shared__ __align__(16) u16 Bs[64][40];    // [n][k]
    const int tid  = threadIdx.x;
    const int wave = tid >> 5, lane = tid & 31;
    const int hl   = lane >> 4, col = lane & 15;
    const int rowblk = blockIdx.y * 128, colblk = blockIdx.x * 64;
    const int wrow = (wave >> 1) * 32, wcol = (wave & 1) * 32;

    // per-thread staging coordinates
    const int ar_row = tid >> 1;              // 0..127
    const int ar_seg = (tid & 1) * 16;        // 0 or 16 (u16)
    const int bn_row = tid >> 2;              // 0..63
    const int bn_seg = (tid & 3) * 8;         // 0,8,16,24 (u16)
    const u16* srcA = A  + (size_t)(rowblk + ar_row) * K + ar_seg;
    const u16* srcB = Bt + (size_t)(colblk + bn_row) * K + bn_seg;

    v8f acc[2][2];
#pragma unroll
    for (int mt = 0; mt < 2; ++mt)
#pragma unroll
        for (int nt = 0; nt < 2; ++nt) acc[mt][nt] = vzero8();

    for (int k0 = 0; k0 < K; k0 += 32) {
#if HAVE_ASYNC_LDS
        async_copy16(srcA + k0,      &As[ar_row][ar_seg]);
        async_copy16(srcA + k0 + 8,  &As[ar_row][ar_seg + 8]);
        async_copy16(srcB + k0,      &Bs[bn_row][bn_seg]);
#else
        {
            const u32* s = (const u32*)(srcA + k0);
            u32* d = (u32*)&As[ar_row][ar_seg];
#pragma unroll
            for (int j = 0; j < 8; ++j) d[j] = s[j];
        }
        {
            const u32* s = (const u32*)(srcB + k0);
            u32* d = (u32*)&Bs[bn_row][bn_seg];
#pragma unroll
            for (int j = 0; j < 4; ++j) d[j] = s[j];
        }
#endif
        if (k0 + 32 < K) {                     // prefetch next k-slab
            __builtin_prefetch(srcA + k0 + 32, 0, 0);
            __builtin_prefetch(srcB + k0 + 32, 0, 0);
        }
#if HAVE_ASYNC_LDS
        wait_async0();
#endif
        __syncthreads();

        Frag af[2], bfr[2];
#pragma unroll
        for (int mt = 0; mt < 2; ++mt) {
            const u16* ar = &As[wrow + mt * 16 + col][0];
#pragma unroll
            for (int j = 0; j < 4; ++j) af[mt].u[j]     = *(const u32*)(ar + hl * 8 + 2 * j);
#pragma unroll
            for (int j = 0; j < 4; ++j) af[mt].u[4 + j] = *(const u32*)(ar + hl * 8 + 16 + 2 * j);
        }
#pragma unroll
        for (int nt = 0; nt < 2; ++nt) {
            const u16* br = &Bs[wcol + nt * 16 + col][0];
#pragma unroll
            for (int j = 0; j < 8; ++j) bfr[nt].u[j] = *(const u32*)(br + hl * 8 + 2 * j + (j >> 2) * 8);
        }
#pragma unroll
        for (int mt = 0; mt < 2; ++mt)
#pragma unroll
            for (int nt = 0; nt < 2; ++nt)
                acc[mt][nt] = wmma_bf16(af[mt].v, bfr[nt].v, acc[mt][nt]);
        __syncthreads();
    }

#pragma unroll
    for (int mt = 0; mt < 2; ++mt)
#pragma unroll
        for (int nt = 0; nt < 2; ++nt)
#pragma unroll
            for (int r = 0; r < 8; ++r)
                C[(size_t)(rowblk + wrow + mt * 16 + hl * 8 + r) * N +
                  colblk + wcol + nt * 16 + col] = acc[mt][nt][r];
}

// ---------------------------------------------------------------------------
// Build per-head Q: [H][T][192] bf16, interleaved RoPE on dims 128..191,
// with 1/sqrt(192) folded in.
// ---------------------------------------------------------------------------
__global__ void k_build_q(const float* __restrict__ q, u16* __restrict__ qfull) {
    int idx = blockIdx.x * 256 + threadIdx.x;
    const int total = NHEADS * T_SEQ * QKD;
    if (idx >= total) return;
    int h   = idx / (T_SEQ * QKD);
    int rem = idx % (T_SEQ * QKD);
    int t = rem / QKD, d = rem % QKD;
    const float scale = 0.07216878364870322f;   // 192^-0.5
    const float* qrow = q + (size_t)t * (NHEADS * QKD) + h * QKD;
    float val;
    if (d < NOPE) {
        val = qrow[d];
    } else {
        int i = (d - NOPE) >> 1, w = (d - NOPE) & 1;
        float inv = __powf(10000.0f, -(float)i / 32.0f);
        float s, c;
        __sincosf((float)t * inv, &s, &c);
        float x1 = qrow[NOPE + 2 * i], x2 = qrow[NOPE + 2 * i + 1];
        val = w ? (x1 * s + x2 * c) : (x1 * c - x2 * s);
    }
    qfull[idx] = f2bf(val * scale);
}

// ---------------------------------------------------------------------------
// Build per-head K: [H][T][192] bf16; nope from kv, rope from kv_a (broadcast).
// ---------------------------------------------------------------------------
__global__ void k_build_k(const float* __restrict__ kv, const float* __restrict__ kv_a,
                          u16* __restrict__ kfull) {
    int idx = blockIdx.x * 256 + threadIdx.x;
    const int total = NHEADS * T_SEQ * QKD;
    if (idx >= total) return;
    int h   = idx / (T_SEQ * QKD);
    int rem = idx % (T_SEQ * QKD);
    int t = rem / QKD, d = rem % QKD;
    float val;
    if (d < NOPE) {
        val = kv[(size_t)t * (NHEADS * (NOPE + V_DIM)) + h * (NOPE + V_DIM) + d];
    } else {
        int i = (d - NOPE) >> 1, w = (d - NOPE) & 1;
        const float* base = kv_a + (size_t)t * (KV_LORA + ROPE_D) + KV_LORA;
        float inv = __powf(10000.0f, -(float)i / 32.0f);
        float s, c;
        __sincosf((float)t * inv, &s, &c);
        float x1 = base[2 * i], x2 = base[2 * i + 1];
        val = w ? (x1 * s + x2 * c) : (x1 * c - x2 * s);
    }
    kfull[idx] = f2bf(val);
}

// ---------------------------------------------------------------------------
// Build transposed per-head V: vt[h][d][t] = kv[t, h*256 + 128 + d], bf16.
// ---------------------------------------------------------------------------
__global__ void k_build_vt(const float* __restrict__ kv, u16* __restrict__ vt) {
    int idx = blockIdx.x * 256 + threadIdx.x;
    const int total = NHEADS * V_DIM * T_SEQ;
    if (idx >= total) return;
    int h   = idx / (V_DIM * T_SEQ);
    int rem = idx % (V_DIM * T_SEQ);
    int d = rem / T_SEQ, t = rem % T_SEQ;
    vt[idx] = f2bf(kv[(size_t)t * (NHEADS * (NOPE + V_DIM)) + h * (NOPE + V_DIM) + NOPE + d]);
}

// ---------------------------------------------------------------------------
// Fused causal flash attention, bf16 WMMA, fp32 online softmax.
// grid = (T/64, NHEADS), block = 128 (4 waves); each wave owns 16 query rows.
// Q already scaled by 1/sqrt(192).
// ---------------------------------------------------------------------------
__global__ __launch_bounds__(128) void k_flash_attn(
    const u16* __restrict__ qfull, const u16* __restrict__ kfull,
    const u16* __restrict__ vt, u16* __restrict__ attn) {
    __shared__ __align__(16) u16 Plds[4][16][34];
    const int wave = threadIdx.x >> 5, lane = threadIdx.x & 31;
    const int hl = lane >> 4, col = lane & 15;
    const int head  = blockIdx.y;
    const int qbase = blockIdx.x * 64 + wave * 16;
    const u16* Qh = qfull + (size_t)head * T_SEQ * QKD;
    const u16* Kh = kfull + (size_t)head * T_SEQ * QKD;
    const u16* Vh = vt    + (size_t)head * V_DIM * T_SEQ;

    // Q fragments (A layout: lane holds row col; half selects k groups)
    Frag qf[6];
    {
        const u16* qrow = Qh + (size_t)(qbase + col) * QKD;
#pragma unroll
        for (int s = 0; s < 6; ++s) {
            int kb = s * 32 + hl * 8;
#pragma unroll
            for (int j = 0; j < 4; ++j) qf[s].u[j]     = *(const u32*)(qrow + kb + 2 * j);
#pragma unroll
            for (int j = 0; j < 4; ++j) qf[s].u[4 + j] = *(const u32*)(qrow + kb + 16 + 2 * j);
        }
    }

    v8f O[8];
#pragma unroll
    for (int v = 0; v < 8; ++v) O[v] = vzero8();
    float mrow[8], lrow[8];
#pragma unroll
    for (int r = 0; r < 8; ++r) { mrow[r] = -3.0e38f; lrow[r] = 0.0f; }

    const int kmax = blockIdx.x * 64 + 63;   // uniform across block (barrier-safe)
    for (int kb = 0; kb <= kmax; kb += 32) {
        v8f S0 = vzero8(), S1 = vzero8();
        const u16* kr0 = Kh + (size_t)(kb + col) * QKD;
        const u16* kr1 = Kh + (size_t)(kb + 16 + col) * QKD;
#pragma unroll
        for (int s = 0; s < 6; ++s) {
            Frag kf0, kf1;
            int off = s * 32 + hl * 8;
#pragma unroll
            for (int j = 0; j < 8; ++j) kf0.u[j] = *(const u32*)(kr0 + off + 2 * j + (j >> 2) * 8);
#pragma unroll
            for (int j = 0; j < 8; ++j) kf1.u[j] = *(const u32*)(kr1 + off + 2 * j + (j >> 2) * 8);
            S0 = wmma_bf16(qf[s].v, kf0.v, S0);
            S1 = wmma_bf16(qf[s].v, kf1.v, S1);
        }
        const int kp0 = kb + col, kp1 = kb + 16 + col;
#pragma unroll
        for (int r = 0; r < 8; ++r) {
            int qp = qbase + hl * 8 + r;
            float s0 = (kp0 <= qp) ? S0[r] : -1.0e30f;
            float s1 = (kp1 <= qp) ? S1[r] : -1.0e30f;
            float mx = fmaxf(s0, s1);
#pragma unroll
            for (int m = 1; m <= 8; m <<= 1) mx = fmaxf(mx, __shfl_xor(mx, m, 32));
            float mnew = fmaxf(mrow[r], mx);
            float corr = __expf(mrow[r] - mnew);
            float p0 = __expf(s0 - mnew);
            float p1 = __expf(s1 - mnew);
            float ps = p0 + p1;
#pragma unroll
            for (int m = 1; m <= 8; m <<= 1) ps += __shfl_xor(ps, m, 32);
            lrow[r] = lrow[r] * corr + ps;
            mrow[r] = mnew;
#pragma unroll
            for (int v = 0; v < 8; ++v) O[v][r] *= corr;
            Plds[wave][hl * 8 + r][col]      = f2bf(p0);
            Plds[wave][hl * 8 + r][16 + col] = f2bf(p1);
        }
        __syncthreads();
        Frag pf;   // re-read P in A-fragment layout
        {
            const u16* prow = &Plds[wave][col][0];
#pragma unroll
            for (int j = 0; j < 4; ++j) pf.u[j]     = *(const u32*)(prow + hl * 8 + 2 * j);
#pragma unroll
            for (int j = 0; j < 4; ++j) pf.u[4 + j] = *(const u32*)(prow + hl * 8 + 16 + 2 * j);
        }
        __syncthreads();
#pragma unroll
        for (int v = 0; v < 8; ++v) {
            Frag vf;
            const u16* vp = Vh + (size_t)(v * 16 + col) * T_SEQ + kb + hl * 8;
#pragma unroll
            for (int j = 0; j < 8; ++j) vf.u[j] = *(const u32*)(vp + 2 * j + (j >> 2) * 8);
            O[v] = wmma_bf16(pf.v, vf.v, O[v]);
        }
    }

#pragma unroll
    for (int v = 0; v < 8; ++v)
#pragma unroll
        for (int r = 0; r < 8; ++r) {
            float o = O[v][r] / lrow[r];
            int row = qbase + hl * 8 + r;
            attn[(size_t)row * (NHEADS * V_DIM) + head * V_DIM + v * 16 + col] = f2bf(o);
        }
}

// ---------------------------------------------------------------------------
extern "C" void kernel_launch(void* const* d_in, const int* in_sizes, int n_in,
                              void* d_out, int out_size, void* d_ws, size_t ws_size,
                              hipStream_t stream) {
    (void)in_sizes; (void)n_in; (void)out_size; (void)ws_size;
    const float* hidden    = (const float*)d_in[1];
    const float* w_q_a     = (const float*)d_in[2];
    const float* q_a_norm  = (const float*)d_in[3];
    const float* w_q_b     = (const float*)d_in[4];
    const float* w_kv_a    = (const float*)d_in[5];
    const float* kv_a_norm = (const float*)d_in[6];
    const float* w_kv_b    = (const float*)d_in[7];
    const float* w_o       = (const float*)d_in[8];
    float* out = (float*)d_out;

    char* ws = (char*)d_ws;
    size_t off = 0;
    auto alloc = [&](size_t bytes) -> void* {
        void* p = ws + off;
        off = (off + bytes + 255) & ~(size_t)255;
        return p;
    };

    const int KVD = NOPE + V_DIM;          // 256
    u16*   Xb    = (u16*)alloc((size_t)T_SEQ * HIDDEN * 2);
    u16*   WqAt  = (u16*)alloc((size_t)HIDDEN * Q_LORA * 2);           // [Q_LORA][HIDDEN]
    u16*   WqBt  = (u16*)alloc((size_t)Q_LORA * NHEADS * QKD * 2);     // [6144][Q_LORA]
    u16*   WkvAt = (u16*)alloc((size_t)HIDDEN * (KV_LORA + ROPE_D) * 2);
    u16*   WkvBt = (u16*)alloc((size_t)KV_LORA * NHEADS * KVD * 2);
    u16*   Wot   = (u16*)alloc((size_t)NHEADS * V_DIM * HIDDEN * 2);   // [HIDDEN][4096]
    float* qa    = (float*)alloc((size_t)T_SEQ * Q_LORA * 4);
    u16*   qcb   = (u16*)alloc((size_t)T_SEQ * Q_LORA * 2);
    float* qm    = (float*)alloc((size_t)T_SEQ * NHEADS * QKD * 4);
    float* kva   = (float*)alloc((size_t)T_SEQ * (KV_LORA + ROPE_D) * 4);
    u16*   compb = (u16*)alloc((size_t)T_SEQ * KV_LORA * 2);
    float* kvm   = (float*)alloc((size_t)T_SEQ * NHEADS * KVD * 4);
    u16*   qfull = (u16*)alloc((size_t)NHEADS * T_SEQ * QKD * 2);
    u16*   kfull = (u16*)alloc((size_t)NHEADS * T_SEQ * QKD * 2);
    u16*   vtb   = (u16*)alloc((size_t)NHEADS * V_DIM * T_SEQ * 2);
    u16*   attnb = (u16*)alloc((size_t)T_SEQ * NHEADS * V_DIM * 2);

    // activations: plain convert
    {
        size_t n = (size_t)T_SEQ * HIDDEN;
        k_cvt_bf16<<<dim3((unsigned)((n + 255) / 256)), 256, 0, stream>>>(hidden, Xb, (int)n);
    }
    // weights: convert + transpose to [N][K]
    auto cvt_t = [&](const float* in, u16* o, int K, int N) {
        size_t n = (size_t)K * N;
        k_cvt_bf16_t<<<dim3((unsigned)((n + 255) / 256)), 256, 0, stream>>>(in, o, K, N);
    };
    cvt_t(w_q_a,  WqAt,  HIDDEN,  Q_LORA);
    cvt_t(w_q_b,  WqBt,  Q_LORA,  NHEADS * QKD);
    cvt_t(w_kv_a, WkvAt, HIDDEN,  KV_LORA + ROPE_D);
    cvt_t(w_kv_b, WkvBt, KV_LORA, NHEADS * KVD);
    cvt_t(w_o,    Wot,   NHEADS * V_DIM, HIDDEN);

    // q_a = hidden @ w_q_a ; rmsnorm -> bf16
    k_gemm_bf16<<<dim3(Q_LORA / 64, T_SEQ / 128), 256, 0, stream>>>(
        Xb, WqAt, qa, T_SEQ, Q_LORA, HIDDEN);
    k_rmsnorm_bf16<<<T_SEQ, 256, 0, stream>>>(qa, Q_LORA, q_a_norm, Q_LORA, qcb, Q_LORA);
    // q = q_c @ w_q_b
    k_gemm_bf16<<<dim3((NHEADS * QKD) / 64, T_SEQ / 128), 256, 0, stream>>>(
        qcb, WqBt, qm, T_SEQ, NHEADS * QKD, Q_LORA);
    // kv_a = hidden @ w_kv_a ; rmsnorm first 512 -> bf16
    k_gemm_bf16<<<dim3((KV_LORA + ROPE_D) / 64, T_SEQ / 128), 256, 0, stream>>>(
        Xb, WkvAt, kva, T_SEQ, KV_LORA + ROPE_D, HIDDEN);
    k_rmsnorm_bf16<<<T_SEQ, 256, 0, stream>>>(kva, KV_LORA + ROPE_D, kv_a_norm, KV_LORA,
                                              compb, KV_LORA);
    // kv = compressed @ w_kv_b
    k_gemm_bf16<<<dim3((NHEADS * KVD) / 64, T_SEQ / 128), 256, 0, stream>>>(
        compb, WkvBt, kvm, T_SEQ, NHEADS * KVD, KV_LORA);

    // assemble per-head Q/K/V (RoPE + scale folded into Q)
    {
        int n = NHEADS * T_SEQ * QKD;
        k_build_q<<<dim3((n + 255) / 256), 256, 0, stream>>>(qm, qfull);
        k_build_k<<<dim3((n + 255) / 256), 256, 0, stream>>>(kvm, kva, kfull);
        int nv = NHEADS * V_DIM * T_SEQ;
        k_build_vt<<<dim3((nv + 255) / 256), 256, 0, stream>>>(kvm, vtb);
    }

    // fused causal attention
    k_flash_attn<<<dim3(T_SEQ / 64, NHEADS), 128, 0, stream>>>(qfull, kfull, vtb, attnb);

    // out = attn @ w_o  (fp32 result straight to d_out)
    k_gemm_bf16<<<dim3(HIDDEN / 64, T_SEQ / 128), 256, 0, stream>>>(
        attnb, Wot, out, T_SEQ, HIDDEN, NHEADS * V_DIM);
}
